// GRU_6605659701958
// MI455X (gfx1250) — matrix operand order repaired and checked by reading, not compiled, not run on hardware
//
#include <hip/hip_runtime.h>
#include <hip/hip_bf16.h>
#include <math.h>

// GRU: S=2048, B=64, I=H=256, fp32 in/out.
// Phase 0: one-time fp32->bf16 conversion of all 6 weight matrices into ws.
// Phase 1: gates[g][s*B+b][h] = x @ Wx[g]^T + bx[g]   (memory-bound, all WGPs)
// Phase 2: single-workgroup persistent scan, recurrent weights pinned in VGPRs
//          as bf16 WMMA fragments, fp32 accumulate + fp32 carried state.

#define S_LEN 2048
#define B_DIM 64
#define I_DIM 256
#define H_DIM 256

typedef __attribute__((ext_vector_type(16))) __bf16 v16bf;
typedef __attribute__((ext_vector_type(8)))  __bf16 v8bf;
typedef __attribute__((ext_vector_type(8)))  float  v8f;

__device__ __forceinline__ v8f wmma_bf16(v16bf a, v16bf b, v8f c) {
  // v_wmma_f32_16x16x32_bf16  D = A(16x32) * B(32x16) + C(16x16 f32)
  return __builtin_amdgcn_wmma_f32_16x16x32_bf16(
      /*neg_a=*/false, a, /*neg_b=*/false, b,
      /*c_mod=*/(short)0, c, /*reuse_a=*/false, /*reuse_b=*/false);
}

// B-matrix fragment from pre-converted bf16: lane L holds column n = L%16,
// 16 contiguous K values at K = ks*32 + (L>=16 ? 16 : 0).  32B-aligned
// vector load -> two global_load_b128, no VALU.
__device__ __forceinline__ v16bf frag_b_from_bf16(const __bf16* p) {
  return *(const v16bf*)p;
}

// A-matrix fragment: lane L holds row m = L%16; 16 contiguous K values at
// ks*32 + koff (koff = 0 or 16). fp32 source.
__device__ __forceinline__ v16bf frag_a_from_f32(const float* __restrict__ p) {
  v16bf r;
#pragma unroll
  for (int i = 0; i < 8; ++i) { r[i] = (__bf16)p[i]; r[8 + i] = (__bf16)p[8 + i]; }
  return r;
}

// Same A layout from an LDS bf16 buffer (two ds_load_b128).
__device__ __forceinline__ v16bf frag_a_from_lds(const __bf16* p) {
  v16bf r;
#pragma unroll
  for (int i = 0; i < 8; ++i) { r[i] = p[i]; r[8 + i] = p[8 + i]; }
  return r;
}

__device__ __forceinline__ float sigmoidf_(float x) {
  return 1.0f / (1.0f + __expf(-x));
}

// ---------------------------------------------------------------------------
// Phase 0: fp32 -> bf16 weight conversion (one 256x256 matrix per launch).
// 32 blocks x 256 threads x 8 elements.
// ---------------------------------------------------------------------------
__global__ __launch_bounds__(256) void cvt_w_bf16(const float* __restrict__ src,
                                                  __bf16* __restrict__ dst) {
  const int i = (blockIdx.x * 256 + threadIdx.x) * 8;
  const float4 a = *(const float4*)(src + i);
  const float4 b = *(const float4*)(src + i + 4);
  v8bf r;
  r[0] = (__bf16)a.x; r[1] = (__bf16)a.y; r[2] = (__bf16)a.z; r[3] = (__bf16)a.w;
  r[4] = (__bf16)b.x; r[5] = (__bf16)b.y; r[6] = (__bf16)b.z; r[7] = (__bf16)b.w;
  *(v8bf*)(dst + i) = r;   // 16B store
}

// ---------------------------------------------------------------------------
// Phase 1 helper: one gate's projection for one 16-row M-tile (A resident).
// Two n-tiles per iteration -> two independent WMMA chains interleave.
// ---------------------------------------------------------------------------
__device__ __forceinline__ void xproj_gate(
    const v16bf (&afrag)[8],
    const __bf16* __restrict__ Wg, const float* __restrict__ bg,
    float* __restrict__ G,
    int m0, int lm, int kbb, int hi8) {
  for (int nt = 0; nt < H_DIM / 16; nt += 2) {
    const int n0 = nt * 16;
    const int n1 = n0 + 16;
    const float bv0 = bg[n0 + lm];
    const float bv1 = bg[n1 + lm];
    v8f acc0, acc1;
#pragma unroll
    for (int v = 0; v < 8; ++v) { acc0[v] = bv0; acc1[v] = bv1; }
    const __bf16* brow0 = Wg + (size_t)(n0 + lm) * I_DIM + kbb;
    const __bf16* brow1 = Wg + (size_t)(n1 + lm) * I_DIM + kbb;
#pragma unroll
    for (int ks = 0; ks < I_DIM / 32; ++ks) {
      v16bf b0 = frag_b_from_bf16(brow0 + ks * 32);
      v16bf b1 = frag_b_from_bf16(brow1 + ks * 32);
      acc0 = wmma_bf16(afrag[ks], b0, acc0);
      acc1 = wmma_bf16(afrag[ks], b1, acc1);
    }
#pragma unroll
    for (int v = 0; v < 8; ++v) {
      G[(size_t)(m0 + hi8 + v) * H_DIM + n0 + lm] = acc0[v];
      G[(size_t)(m0 + hi8 + v) * H_DIM + n1 + lm] = acc1[v];
    }
  }
}

// ---------------------------------------------------------------------------
// Phase 1: x projections. One wave per 16-row M-tile (M = S*B = 131072).
// 256 threads = 8 waves; grid = 1024 blocks.  x is read from HBM exactly
// once: the 16x256 A-strip lives in 64 VGPRs across all 3 gates/16 n-tiles.
// ---------------------------------------------------------------------------
__global__ __launch_bounds__(256) void gru_xproj(
    const float* __restrict__ x,
    const __bf16* __restrict__ Wxz_b, const __bf16* __restrict__ Wxr_b,
    const __bf16* __restrict__ Wxh_b,
    const float* __restrict__ bxz, const float* __restrict__ bxr,
    const float* __restrict__ bxh,
    float* __restrict__ gates) {
  const int lane = threadIdx.x & 31;
  const int wave = threadIdx.x >> 5;
  const int mt   = blockIdx.x * 8 + wave;   // 16-row tile index, 0..8191
  const int m0   = mt * 16;
  const int lm   = lane & 15;
  const int koff = (lane >> 4) << 4;        // A-frag chunk offset (0 / 16)
  const int kbb  = (lane >> 4) << 4;        // B-frag K offset (0 / 16)
  const int hi8  = (lane >> 4) << 3;        // C/D row offset (0 / 8)
  const size_t SBH = (size_t)S_LEN * B_DIM * H_DIM;

  const float* arow = x + (size_t)(m0 + lm) * I_DIM;
  v16bf afrag[8];
#pragma unroll
  for (int ks = 0; ks < 8; ++ks)
    afrag[ks] = frag_a_from_f32(arow + ks * 32 + koff);

  xproj_gate(afrag, Wxz_b, bxz, gates,           m0, lm, kbb, hi8);
  xproj_gate(afrag, Wxr_b, bxr, gates + SBH,     m0, lm, kbb, hi8);
  xproj_gate(afrag, Wxh_b, bxh, gates + 2 * SBH, m0, lm, kbb, hi8);
}

// ---------------------------------------------------------------------------
// Phase 2: persistent single-workgroup scan. 512 threads = 16 waves; wave w
// owns output columns [16w, 16w+16) and holds the corresponding bf16
// B-fragments of Whz/Whr/Whh (3 gates x 8 ksteps x 8 VGPRs = 192 VGPRs).
// ---------------------------------------------------------------------------
__global__ __launch_bounds__(512, 1) void gru_scan(
    const float* __restrict__ h0,
    const __bf16* __restrict__ Whz_b, const __bf16* __restrict__ Whr_b,
    const __bf16* __restrict__ Whh_b,
    const float* __restrict__ bhz, const float* __restrict__ bhr,
    const float* __restrict__ bhh,
    const float* __restrict__ gates,
    float* __restrict__ out) {
  __shared__ float  hbuf [B_DIM * H_DIM];   // fp32 carried state      64 KB
  __shared__ __bf16 hb16 [B_DIM * H_DIM];   // bf16 mirror (WMMA A)    32 KB
  __shared__ __bf16 rhb16[B_DIM * H_DIM];   // r * h (WMMA A)          32 KB
  __shared__ __bf16 zb16 [B_DIM * H_DIM];   // update gate z           32 KB

  const int tid   = threadIdx.x;
  const int lane  = tid & 31;
  const int wave  = tid >> 5;               // 0..15 -> n-tile
  const int lm    = lane & 15;
  const int hi8   = (lane >> 4) << 3;
  const int koffA = (lane >> 4) << 4;
  const int kbB   = (lane >> 4) << 4;
  const int nn    = wave * 16 + lm;

  // init carried state
  for (int i = tid; i < B_DIM * H_DIM; i += 512) {
    float h = h0[i];
    hbuf[i] = h;
    hb16[i] = (__bf16)h;
  }

  // Pin recurrent weights in VGPRs (bf16, pre-converted: direct vector loads).
  v16bf wf0[8], wf1[8], wf2[8];
  {
    const __bf16* rz = Whz_b + (size_t)nn * H_DIM + kbB;
    const __bf16* rr = Whr_b + (size_t)nn * H_DIM + kbB;
    const __bf16* rh = Whh_b + (size_t)nn * H_DIM + kbB;
#pragma unroll
    for (int ks = 0; ks < 8; ++ks) {
      wf0[ks] = frag_b_from_bf16(rz + ks * 32);
      wf1[ks] = frag_b_from_bf16(rr + ks * 32);
      wf2[ks] = frag_b_from_bf16(rh + ks * 32);
    }
  }
  const float bz = bhz[nn], br = bhr[nn], bh = bhh[nn];

  const size_t SBH = (size_t)S_LEN * B_DIM * H_DIM;

  __syncthreads();

  for (int s = 0; s < S_LEN; ++s) {
    const float* gzs = gates           + (size_t)s * (B_DIM * H_DIM);
    const float* grs = gates + SBH     + (size_t)s * (B_DIM * H_DIM);
    const float* ghs = gates + 2 * SBH + (size_t)s * (B_DIM * H_DIM);
    float* outs = out + (size_t)s * (B_DIM * H_DIM);

    // Prefetch next step's gate rows into L2/WGP$ (global_prefetch_b8).
    if (s + 1 < S_LEN) {
      const int r_ = tid >> 3;              // 64 rows
      const int c_ = (tid & 7) * 32;        // 8 x 128B chunks per 1KB row
      const size_t off = (size_t)(B_DIM + r_) * H_DIM + c_;
      __builtin_prefetch(gzs + off, 0, 1);
      __builtin_prefetch(grs + off, 0, 1);
      __builtin_prefetch(ghs + off, 0, 1);
    }

    // ---- stage 1: z and r gates (A-frag of h shared between both GEMMs,
    //      two independent WMMA chains interleave to cover D->C hazards) ----
#pragma unroll
    for (int mtc = 0; mtc < 4; ++mtc) {
      const int bm = mtc * 16;
      v8f az, ar;
#pragma unroll
      for (int v = 0; v < 8; ++v) {
        const int gi = (bm + hi8 + v) * H_DIM + nn;
        az[v] = gzs[gi] + bz;
        ar[v] = grs[gi] + br;
      }
#pragma unroll
      for (int ks = 0; ks < 8; ++ks) {
        v16bf a = frag_a_from_lds(hb16 + (bm + lm) * H_DIM + ks * 32 + koffA);
        az = wmma_bf16(a, wf0[ks], az);
        ar = wmma_bf16(a, wf1[ks], ar);
      }
#pragma unroll
      for (int v = 0; v < 8; ++v) {
        const int idx = (bm + hi8 + v) * H_DIM + nn;
        zb16[idx]  = (__bf16)sigmoidf_(az[v]);
        rhb16[idx] = (__bf16)(sigmoidf_(ar[v]) * hbuf[idx]);
      }
    }
    __syncthreads();

    // ---- stage 2: candidate + update; two m-tiles processed concurrently ----
#pragma unroll
    for (int mp = 0; mp < 2; ++mp) {
      const int bmA = mp * 32;
      const int bmB = bmA + 16;
      v8f ahA, ahB;
#pragma unroll
      for (int v = 0; v < 8; ++v) {
        ahA[v] = ghs[(bmA + hi8 + v) * H_DIM + nn] + bh;
        ahB[v] = ghs[(bmB + hi8 + v) * H_DIM + nn] + bh;
      }
#pragma unroll
      for (int ks = 0; ks < 8; ++ks) {
        v16bf aA = frag_a_from_lds(rhb16 + (bmA + lm) * H_DIM + ks * 32 + koffA);
        v16bf aB = frag_a_from_lds(rhb16 + (bmB + lm) * H_DIM + ks * 32 + koffA);
        ahA = wmma_bf16(aA, wf2[ks], ahA);
        ahB = wmma_bf16(aB, wf2[ks], ahB);
      }
#pragma unroll
      for (int v = 0; v < 8; ++v) {
        {
          const int idx = (bmA + hi8 + v) * H_DIM + nn;
          const float hc = tanhf(ahA[v]);
          const float zv = (float)zb16[idx];
          const float hn = (1.0f - zv) * hbuf[idx] + zv * hc;
          hbuf[idx] = hn; hb16[idx] = (__bf16)hn; outs[idx] = hn;
        }
        {
          const int idx = (bmB + hi8 + v) * H_DIM + nn;
          const float hc = tanhf(ahB[v]);
          const float zv = (float)zb16[idx];
          const float hn = (1.0f - zv) * hbuf[idx] + zv * hc;
          hbuf[idx] = hn; hb16[idx] = (__bf16)hn; outs[idx] = hn;
        }
      }
    }
    __syncthreads();
  }

  // h_last appended after outputs
  for (int i = tid; i < B_DIM * H_DIM; i += 512)
    out[SBH + i] = hbuf[i];
}

// ---------------------------------------------------------------------------
extern "C" void kernel_launch(void* const* d_in, const int* in_sizes, int n_in,
                              void* d_out, int out_size, void* d_ws, size_t ws_size,
                              hipStream_t stream) {
  (void)in_sizes; (void)n_in; (void)out_size; (void)ws_size;
  const float* x   = (const float*)d_in[0];
  const float* h0  = (const float*)d_in[1];
  const float* Wxz = (const float*)d_in[2];
  const float* Whz = (const float*)d_in[3];
  const float* Wxr = (const float*)d_in[4];
  const float* Whr = (const float*)d_in[5];
  const float* Wxh = (const float*)d_in[6];
  const float* Whh = (const float*)d_in[7];
  const float* bxz = (const float*)d_in[8];
  const float* bhz = (const float*)d_in[9];
  const float* bxr = (const float*)d_in[10];
  const float* bhr = (const float*)d_in[11];
  const float* bxh = (const float*)d_in[12];
  const float* bhh = (const float*)d_in[13];
  float* out   = (float*)d_out;
  float* gates = (float*)d_ws;   // 3*S*B*H fp32 = 384 MiB

  // bf16 weight copies live after the gate buffers in ws (+0.75 MiB).
  const size_t SBH = (size_t)S_LEN * B_DIM * H_DIM;
  const size_t WN  = (size_t)H_DIM * I_DIM;          // 65536 elements
  __bf16* wb = (__bf16*)((char*)d_ws + 3 * SBH * sizeof(float));
  __bf16 *Wxz_b = wb,          *Wxr_b = wb + WN,     *Wxh_b = wb + 2 * WN;
  __bf16 *Whz_b = wb + 3 * WN, *Whr_b = wb + 4 * WN, *Whh_b = wb + 5 * WN;

  // Phase 0: weight conversion (32 blocks x 256 thr x 8 elem = 65536 each).
  cvt_w_bf16<<<dim3(32), 256, 0, stream>>>(Wxz, Wxz_b);
  cvt_w_bf16<<<dim3(32), 256, 0, stream>>>(Wxr, Wxr_b);
  cvt_w_bf16<<<dim3(32), 256, 0, stream>>>(Wxh, Wxh_b);
  cvt_w_bf16<<<dim3(32), 256, 0, stream>>>(Whz, Whz_b);
  cvt_w_bf16<<<dim3(32), 256, 0, stream>>>(Whr, Whr_b);
  cvt_w_bf16<<<dim3(32), 256, 0, stream>>>(Whh, Whh_b);

  // Phase 1: 8192 M-tiles, 8 waves/block -> 1024 blocks.
  gru_xproj<<<dim3((S_LEN * B_DIM) / (16 * 8)), 256, 0, stream>>>(
      x, Wxz_b, Wxr_b, Wxh_b, bxz, bxr, bxh, gates);

  // Phase 2: persistent single-WGP scan.
  gru_scan<<<dim3(1), 512, 0, stream>>>(
      h0, Whz_b, Whr_b, Whh_b, bhz, bhr, bhh, gates, out);
}
